// TransformerBlock_47210280517804
// MI455X (gfx1250) — matrix operand order repaired and checked by reading, not compiled, not run on hardware
//
#include <hip/hip_runtime.h>

// ---------------------------------------------------------------------------
// Types for CDNA5 WMMA (wave32, 16x16x32 bf16 -> f32) and TDM descriptors
// ---------------------------------------------------------------------------
typedef __bf16 bf16;
typedef __attribute__((ext_vector_type(8)))  __bf16 v8bf;
typedef __attribute__((ext_vector_type(16))) __bf16 v16bf;
typedef __attribute__((ext_vector_type(8)))  float  v8f;
typedef __attribute__((ext_vector_type(4)))  unsigned int u32x4;
typedef __attribute__((ext_vector_type(8)))  int i32x8;
typedef __attribute__((ext_vector_type(4)))  int i32x4;

// Tensor Data Mover availability (arity differs between toolchains)
#if defined(__has_builtin)
#if __has_builtin(__builtin_amdgcn_tensor_load_to_lds) && \
    __has_builtin(__builtin_amdgcn_s_wait_tensorcnt)
#define HAVE_TDM 1
#endif
#endif
#ifndef HAVE_TDM
#define HAVE_TDM 0
#endif

// Problem constants (fixed by the reference)
#define BB   4
#define TT   2048
#define CC   1024
#define HH   16
#define HS   64
#define FF   4096
#define MM   (BB * TT)          // 8192 rows
#define TILE_K 128              // K-chunk staged in LDS (x2 buffers, 64KB total)
#define ATTN_SCALE 0.03125f     // C^-0.5 = 1/32 (reference scales by n_embed!)

// ---------------------------------------------------------------------------
// Fragment helpers
// ---------------------------------------------------------------------------
__device__ __forceinline__ v16bf make_frag(const bf16* p_lo, const bf16* p_hi) {
    v8bf lo = *(const v8bf*)p_lo;
    v8bf hi = *(const v8bf*)p_hi;
    v16bf f;
#pragma unroll
    for (int i = 0; i < 8; ++i) { f[i] = lo[i]; f[i + 8] = hi[i]; }
    return f;
}

__device__ __forceinline__ v8f wmma_bf16(v16bf a, v16bf b, v8f c) {
    return __builtin_amdgcn_wmma_f32_16x16x32_bf16(
        /*neg_a=*/false, a, /*neg_b=*/false, b,
        /*c_mod=*/(short)0, c, /*reuse_a=*/false, /*reuse_b=*/false);
}

// A-matrix 16x32 bf16 fragment: lane (half,lm) holds row m=lm,
//   half==0: K = {k0..k0+7, k0+16..k0+23}, half==1: K = {k0+8.., k0+24..}
__device__ __forceinline__ v16bf load_a_frag(const bf16* row, int k0, int half) {
    return make_frag(row + k0 + half * 8, row + k0 + 16 + half * 8);
}
// B-matrix 32x16 bf16 fragment from Bt[N][K] (row n = output col, contiguous K):
//   lane (half,lm) holds col n=lm, K = {k0+half*8+0..7, k0+16+half*8+0..7}
__device__ __forceinline__ v16bf load_b_frag(const bf16* row, int k0, int half) {
    return make_frag(row + k0 + half * 8, row + k0 + 16 + half * 8);
}

__device__ __forceinline__ float rowmax16(float v) {
#pragma unroll
    for (int m = 1; m < 16; m <<= 1) v = fmaxf(v, __shfl_xor(v, m, 32));
    return v;
}
__device__ __forceinline__ float rowsum16(float v) {
#pragma unroll
    for (int m = 1; m < 16; m <<= 1) v += __shfl_xor(v, m, 32);
    return v;
}

#if HAVE_TDM
// Issue a TDM load of a [16 rows x TILE_K cols] bf16 tile (row stride = K
// elements) from global 'gptr' into per-wave LDS at 'lds_off'.
__device__ __forceinline__ void tdm_load_a_tile(const bf16* gptr, unsigned lds_off,
                                                int K) {
    unsigned long long ga = (unsigned long long)(__UINTPTR_TYPE__)gptr;
    unsigned ga_lo = (unsigned)__builtin_amdgcn_readfirstlane((int)(unsigned)ga);
    unsigned ga_hi = (unsigned)__builtin_amdgcn_readfirstlane((int)(unsigned)(ga >> 32));
    unsigned la    = (unsigned)__builtin_amdgcn_readfirstlane((int)lds_off);

    // D# group 0: count=1 | lds_addr | global_addr[56:0] | type=2
    u32x4 g0 = { 1u, la, ga_lo, (ga_hi & 0x01FFFFFFu) | 0x80000000u };
    // D# group 1: data_size=2B; tensor_dim0=K, tensor_dim1=M;
    //             tile_dim0=TILE_K, tile_dim1=16; tensor_dim0_stride=K
    i32x8 g1 = {
        0x00010000,                                   // data_size = 1 (2 bytes)
        (int)((K & 0xFFFF) << 16),                    // tensor_dim0[15:0]
        (int)(((K >> 16) & 0xFFFF) | ((MM & 0xFFFF) << 16)),
        (int)(((MM >> 16) & 0xFFFF) | (TILE_K << 16)),// tile_dim0
        16,                                           // tile_dim1
        K,                                            // tensor_dim0_stride[31:0]
        0, 0
    };
    i32x4 g2 = {0, 0, 0, 0};
    i32x4 g3 = {0, 0, 0, 0};
#if __clang_major__ >= 23
    i32x8 g4 = {0, 0, 0, 0, 0, 0, 0, 0};
    __builtin_amdgcn_tensor_load_to_lds(g0, g1, g2, g3, g4, 0);
#else
    __builtin_amdgcn_tensor_load_to_lds(g0, g1, g2, g3, 0);
#endif
}
#endif

// ---------------------------------------------------------------------------
// Weight prep: fp32 -> bf16, transposed to [N][K] for contiguous B-fragments
// ---------------------------------------------------------------------------
__global__ void prep_qkv_w(const float* __restrict__ Wq,
                           const float* __restrict__ Wk,
                           const float* __restrict__ Wv,
                           bf16* __restrict__ out) {
    long idx = (long)blockIdx.x * blockDim.x + threadIdx.x;
    if (idx >= 3L * CC * CC) return;
    int n = (int)(idx >> 10);          // / C
    int c = (int)(idx & (CC - 1));
    int sel = n >> 10;                 // n / C
    int rr  = n & (CC - 1);
    int h   = rr >> 6;                 // / HS
    int d   = rr & (HS - 1);
    const float* W = (sel == 0) ? Wq : (sel == 1) ? Wk : Wv;
    out[idx] = (bf16)W[((long)h * CC + c) * HS + d];
}

// out[n*K + k] = (bf16) in[k*N + n];  total K*N elements
__global__ void transpose_w(const float* __restrict__ in, bf16* __restrict__ out,
                            int K, int N) {
    long idx = (long)blockIdx.x * blockDim.x + threadIdx.x;
    if (idx >= (long)K * N) return;
    int n = (int)(idx / K);
    int k = (int)(idx % K);
    out[idx] = (bf16)in[(long)k * N + n];
}

// ---------------------------------------------------------------------------
// LayerNorm: one wave per row of C=1024; fp32 out + bf16 out
// ---------------------------------------------------------------------------
__global__ __launch_bounds__(256) void ln_kernel(const float* __restrict__ x,
                                                 const float* __restrict__ g,
                                                 const float* __restrict__ beta,
                                                 float* __restrict__ yF,
                                                 bf16*  __restrict__ yH,
                                                 int rows) {
    int lane = threadIdx.x & 31, wave = threadIdx.x >> 5;
    long row = (long)blockIdx.x * (blockDim.x >> 5) + wave;
    if (row >= rows) return;
    const float* xr = x + row * CC;
    float vals[32];
    float s = 0.f, s2 = 0.f;
#pragma unroll
    for (int i = 0; i < 32; ++i) {
        float v = xr[i * 32 + lane];
        vals[i] = v; s += v; s2 += v * v;
    }
#pragma unroll
    for (int m = 1; m < 32; m <<= 1) {
        s  += __shfl_xor(s,  m, 32);
        s2 += __shfl_xor(s2, m, 32);
    }
    float mu   = s * (1.f / CC);
    float var  = s2 * (1.f / CC) - mu * mu;
    float rstd = rsqrtf(var + 1e-5f);
#pragma unroll
    for (int i = 0; i < 32; ++i) {
        int c = i * 32 + lane;
        float y = (vals[i] - mu) * rstd * g[c] + beta[c];
        yF[row * CC + c] = y;
        yH[row * CC + c] = (bf16)y;
    }
}

// ---------------------------------------------------------------------------
// bf16 WMMA GEMM, one wave per 16x64 tile (4 accumulators; A reused 4x).
// A-slabs (16 x TILE_K) are staged into per-wave LDS by the Tensor Data Mover
// with double buffering (TENSORcnt in-order per wave: wait<=1 means the older
// buffer is ready). B fragments are preloaded 4-wide per k-step so the
// compiler can clause the 8 b128 loads and use partial loadcnt waits.
//   mode 0: QKV scatter -> qb[B,H,T,HS], kb[B,H,T,HS], vtb[B,H,HS,T]  (bf16)
//   mode 1/3: outF = acc + bias + resid                                (f32)
//   mode 2: outH = relu(acc + bias)                                    (bf16)
// ---------------------------------------------------------------------------
__global__ __launch_bounds__(256) void gemm_bf16_ep(
    const bf16* __restrict__ A, const bf16* __restrict__ Bt,
    int M, int N, int K, long totalTiles, int mode,
    const float* __restrict__ bias, const float* __restrict__ resid,
    float* __restrict__ outF, bf16* __restrict__ outH,
    bf16* __restrict__ qb, bf16* __restrict__ kb, bf16* __restrict__ vtb) {

    int lane = threadIdx.x & 31, wave = threadIdx.x >> 5;
    int lm = lane & 15, half = lane >> 4;
    long tile = (long)blockIdx.x * (blockDim.x >> 5) + wave;
    if (tile >= totalTiles) return;
    int ntn = N >> 6;                        // N / 64
    int mt  = (int)(tile / ntn);
    int nt4 = (int)(tile % ntn);
    int m0 = mt * 16, n0 = nt4 * 64;

    __shared__ bf16 alds_all[8][2][16 * TILE_K];
    bf16* buf0 = alds_all[wave][0];
    bf16* buf1 = alds_all[wave][1];

    const bf16* arow = A + (long)(m0 + lm) * K;   // fallback (no-TDM) path
    const bf16* brow[4];
#pragma unroll
    for (int j = 0; j < 4; ++j)
        brow[j] = Bt + (long)(n0 + j * 16 + lm) * K;

    v8f acc[4];
#pragma unroll
    for (int j = 0; j < 4; ++j) acc[j] = (v8f){0.f,0.f,0.f,0.f,0.f,0.f,0.f,0.f};

#if HAVE_TDM
    tdm_load_a_tile(A + (long)m0 * K,
                    (unsigned)(__UINTPTR_TYPE__)(void*)buf0, K);
#endif
    int nchunks = K / TILE_K;
    for (int c = 0; c < nchunks; ++c) {
        int kc = c * TILE_K;
        bf16* cur = (c & 1) ? buf1 : buf0;
#if HAVE_TDM
        bf16* nxt = (c & 1) ? buf0 : buf1;
        if (c + 1 < nchunks) {
            tdm_load_a_tile(A + (long)m0 * K + kc + TILE_K,
                            (unsigned)(__UINTPTR_TYPE__)(void*)nxt, K);
#pragma unroll
            for (int j = 0; j < 4; ++j)          // hint next B chunk into cache
                __builtin_prefetch(brow[j] + kc + TILE_K, 0, 0);
            __builtin_amdgcn_s_wait_tensorcnt(1);  // older (cur) buffer ready
        } else {
            __builtin_amdgcn_s_wait_tensorcnt(0);
        }
#endif
#pragma unroll
        for (int kk = 0; kk < TILE_K; kk += 32) {
#if HAVE_TDM
            v16bf af = make_frag(cur + lm * TILE_K + kk + half * 8,
                                 cur + lm * TILE_K + kk + 16 + half * 8);
#else
            v16bf af = load_a_frag(arow, kc + kk, half);
#endif
            // Preload all four B fragments so loads clause together and the
            // scheduler can use partial s_wait_loadcnt before each WMMA.
            v16bf b0 = load_b_frag(brow[0], kc + kk, half);
            v16bf b1 = load_b_frag(brow[1], kc + kk, half);
            v16bf b2 = load_b_frag(brow[2], kc + kk, half);
            v16bf b3 = load_b_frag(brow[3], kc + kk, half);
            acc[0] = wmma_bf16(af, b0, acc[0]);
            acc[1] = wmma_bf16(af, b1, acc[1]);
            acc[2] = wmma_bf16(af, b2, acc[2]);
            acc[3] = wmma_bf16(af, b3, acc[3]);
        }
    }

#pragma unroll
    for (int j = 0; j < 4; ++j) {
        int n = n0 + j * 16 + lm;
        float bv = bias ? bias[n] : 0.f;
#pragma unroll
        for (int r = 0; r < 8; ++r) {
            int m = m0 + half * 8 + r;
            float val = acc[j][r] + bv;
            if (mode == 0) {
                int sel = n >> 10;           // 0:q 1:k 2:v
                int rr  = n & (CC - 1);
                int h   = rr >> 6;
                int d   = rr & (HS - 1);
                int b   = m >> 11;           // / T
                int t   = m & (TT - 1);
                if (sel == 0)
                    qb[(((long)b * HH + h) * TT + t) * HS + d] = (bf16)val;
                else if (sel == 1)
                    kb[(((long)b * HH + h) * TT + t) * HS + d] = (bf16)val;
                else
                    vtb[(((long)b * HH + h) * HS + d) * TT + t] = (bf16)val;
            } else if (mode == 2) {
                outH[(long)m * N + n] = (bf16)fmaxf(val, 0.f);
            } else { // 1 and 3: bias + residual, fp32
                long o = (long)m * N + n;
                outF[o] = val + resid[o];
            }
        }
    }
}

// ---------------------------------------------------------------------------
// Flash attention: one wave handles a 16-query tile of one (b,h).
// Scores via 2x WMMA (K=64 -> two K=32 steps), online softmax in fp32,
// P transposed through LDS (C-layout -> A-layout), att@V via 4x WMMA / chunk.
// ---------------------------------------------------------------------------
__global__ __launch_bounds__(256) void attn_kernel(
    const bf16* __restrict__ qb, const bf16* __restrict__ kb,
    const bf16* __restrict__ vtb, bf16* __restrict__ obuf) {

    int lane = threadIdx.x & 31, wave = threadIdx.x >> 5;
    int lm = lane & 15, half = lane >> 4;
    long gtile = (long)blockIdx.x * (blockDim.x >> 5) + wave;
    if (gtile >= (long)BB * HH * (TT / 16)) return;

    int ttile = (int)(gtile % (TT / 16));
    int bh    = (int)(gtile / (TT / 16));
    int t0    = ttile * 16;
    int b     = bh / HH, h = bh % HH;

    const bf16* q  = qb  + (long)bh * TT * HS;
    const bf16* k  = kb  + (long)bh * TT * HS;
    const bf16* vt = vtb + (long)bh * HS * TT;

    __shared__ bf16 plds_all[8][16 * 32];
    bf16* plds = plds_all[wave];

    // Query fragments are invariant over the key loop: K=64 -> two frags.
    const bf16* qrow = q + (long)(t0 + lm) * HS;
    v16bf qa0 = load_a_frag(qrow, 0,  half);
    v16bf qa1 = load_a_frag(qrow, 32, half);

    v8f oacc[4];
#pragma unroll
    for (int dt = 0; dt < 4; ++dt) oacc[dt] = (v8f){0.f,0.f,0.f,0.f,0.f,0.f,0.f,0.f};
    float rmax[8], rsum[8];
#pragma unroll
    for (int r = 0; r < 8; ++r) { rmax[r] = -1e30f; rsum[r] = 0.f; }

    int send = t0 + 16;                       // last key (exclusive)
    for (int s0 = 0; s0 < send; s0 += 32) {
        // --- scores: two 16x16 tiles (keys s0..s0+15, s0+16..s0+31) ---
        const bf16* krA = k + (long)(s0 + lm) * HS;
        const bf16* krB = k + (long)(s0 + 16 + lm) * HS;
        v8f sA = {0.f,0.f,0.f,0.f,0.f,0.f,0.f,0.f};
        v8f sB = {0.f,0.f,0.f,0.f,0.f,0.f,0.f,0.f};
        sA = wmma_bf16(qa0, load_b_frag(krA, 0,  half), sA);
        sA = wmma_bf16(qa1, load_b_frag(krA, 32, half), sA);
        sB = wmma_bf16(qa0, load_b_frag(krB, 0,  half), sB);
        sB = wmma_bf16(qa1, load_b_frag(krB, 32, half), sB);

        // --- causal mask, online softmax update (per-row, fp32) ---
        int sa = s0 + lm, sb = s0 + 16 + lm;
#pragma unroll
        for (int r = 0; r < 8; ++r) {
            int m = t0 + half * 8 + r;
            float a   = sA[r] * ATTN_SCALE; if (sa > m) a   = -1e30f;
            float bbv = sB[r] * ATTN_SCALE; if (sb > m) bbv = -1e30f;
            float mx   = rowmax16(fmaxf(a, bbv));
            float mnew = fmaxf(rmax[r], mx);
            float alpha = __expf(rmax[r] - mnew);
            rmax[r] = mnew;
            float pa = __expf(a - mnew);
            float pb = __expf(bbv - mnew);
            rsum[r] = rsum[r] * alpha + rowsum16(pa + pb);
            oacc[0][r] *= alpha; oacc[1][r] *= alpha;
            oacc[2][r] *= alpha; oacc[3][r] *= alpha;
            int ml = half * 8 + r;                    // local row in tile
            plds[ml * 32 + lm]      = (bf16)pa;       // P[ml][lm]
            plds[ml * 32 + 16 + lm] = (bf16)pb;       // P[ml][16+lm]
        }

        // --- att @ V: P(16x32, from LDS in A-layout) x Vt rows (contig s) ---
        v16bf pfrag = make_frag(plds + lm * 32 + half * 8,
                                plds + lm * 32 + 16 + half * 8);
#pragma unroll
        for (int dt = 0; dt < 4; ++dt) {
            const bf16* vrow = vt + (long)(dt * 16 + lm) * TT + s0;
            oacc[dt] = wmma_bf16(pfrag, load_b_frag(vrow, 0, half), oacc[dt]);
        }
    }

    // --- finalize: divide by row-sums, concat heads into obuf[B,T,C] bf16 ---
    float inv[8];
#pragma unroll
    for (int r = 0; r < 8; ++r) inv[r] = 1.f / rsum[r];
#pragma unroll
    for (int dt = 0; dt < 4; ++dt) {
#pragma unroll
        for (int r = 0; r < 8; ++r) {
            int m = t0 + half * 8 + r;
            int d = h * HS + dt * 16 + lm;
            obuf[((long)b * TT + m) * CC + d] = (bf16)(oacc[dt][r] * inv[r]);
        }
    }
}

// ---------------------------------------------------------------------------
// Host launcher
// ---------------------------------------------------------------------------
extern "C" void kernel_launch(void* const* d_in, const int* in_sizes, int n_in,
                              void* d_out, int out_size, void* d_ws, size_t ws_size,
                              hipStream_t stream) {
    const float* x   = (const float*)d_in[0];
    const float* Wq  = (const float*)d_in[1];
    const float* Wk  = (const float*)d_in[2];
    const float* Wv  = (const float*)d_in[3];
    const float* Wp  = (const float*)d_in[4];
    const float* bp  = (const float*)d_in[5];
    const float* W1  = (const float*)d_in[6];
    const float* b1  = (const float*)d_in[7];
    const float* W2  = (const float*)d_in[8];
    const float* b2  = (const float*)d_in[9];
    const float* g1  = (const float*)d_in[10];
    const float* be1 = (const float*)d_in[11];
    const float* g2  = (const float*)d_in[12];
    const float* be2 = (const float*)d_in[13];

    // Workspace carve-out (256B aligned)
    char* ws = (char*)d_ws;
    auto alloc = [&](size_t bytes) -> char* {
        char* p = ws;
        ws += (bytes + 255) & ~(size_t)255;
        return p;
    };
    float* x1f   = (float*)alloc((size_t)MM * CC * 4);
    float* x2f   = (float*)alloc((size_t)MM * CC * 4);
    float* x3f   = (float*)alloc((size_t)MM * CC * 4);
    bf16*  x1h   = (bf16*) alloc((size_t)MM * CC * 2);
    bf16*  x3h   = (bf16*) alloc((size_t)MM * CC * 2);
    bf16*  qb    = (bf16*) alloc((size_t)MM * CC * 2);
    bf16*  kbuf  = (bf16*) alloc((size_t)MM * CC * 2);
    bf16*  vtb   = (bf16*) alloc((size_t)MM * CC * 2);
    bf16*  obuf  = (bf16*) alloc((size_t)MM * CC * 2);
    bf16*  hb    = (bf16*) alloc((size_t)MM * FF * 2);
    bf16*  Wqkvt = (bf16*) alloc((size_t)3 * CC * CC * 2);
    bf16*  Wpt   = (bf16*) alloc((size_t)CC * CC * 2);
    bf16*  W1t   = (bf16*) alloc((size_t)CC * FF * 2);
    bf16*  W2t   = (bf16*) alloc((size_t)FF * CC * 2);

    auto nb = [](long n, int tpb) { return (unsigned)((n + tpb - 1) / tpb); };

    // 1) weights -> bf16 transposed
    prep_qkv_w <<<nb(3L * CC * CC, 256), 256, 0, stream>>>(Wq, Wk, Wv, Wqkvt);
    transpose_w<<<nb((long)CC * CC, 256), 256, 0, stream>>>(Wp, Wpt, CC, CC);
    transpose_w<<<nb((long)CC * FF, 256), 256, 0, stream>>>(W1, W1t, CC, FF);
    transpose_w<<<nb((long)FF * CC, 256), 256, 0, stream>>>(W2, W2t, FF, CC);

    // 2) LN1
    ln_kernel<<<nb(MM, 8) , 256, 0, stream>>>(x, g1, be1, x1f, x1h, MM);

    // 3) QKV projection (M=8192, N=3072, K=1024) with scatter epilogue
    {
        long tiles = (long)(MM / 16) * (3 * CC / 64);
        gemm_bf16_ep<<<nb(tiles, 8), 256, 0, stream>>>(
            x1h, Wqkvt, MM, 3 * CC, CC, tiles, /*mode=*/0,
            nullptr, nullptr, nullptr, nullptr, qb, kbuf, vtb);
    }

    // 4) causal flash attention -> obuf [B,T,C] bf16
    attn_kernel<<<nb((long)BB * HH * (TT / 16), 8), 256, 0, stream>>>(qb, kbuf, vtb, obuf);

    // 5) output projection + bias + residual(x1): x2 = x1 + o@Wp + bp
    {
        long tiles = (long)(MM / 16) * (CC / 64);
        gemm_bf16_ep<<<nb(tiles, 8), 256, 0, stream>>>(
            obuf, Wpt, MM, CC, CC, tiles, /*mode=*/1,
            bp, x1f, x2f, nullptr, nullptr, nullptr, nullptr);
    }

    // 6) LN2
    ln_kernel<<<nb(MM, 8), 256, 0, stream>>>(x2f, g2, be2, x3f, x3h, MM);

    // 7) FFN1: h = relu(x3 @ W1 + b1), bf16
    {
        long tiles = (long)(MM / 16) * (FF / 64);
        gemm_bf16_ep<<<nb(tiles, 8), 256, 0, stream>>>(
            x3h, W1t, MM, FF, CC, tiles, /*mode=*/2,
            b1, nullptr, nullptr, hb, nullptr, nullptr, nullptr);
    }

    // 8) FFN2 + bias + residual(x3) -> d_out fp32
    {
        long tiles = (long)(MM / 16) * (CC / 64);
        gemm_bf16_ep<<<nb(tiles, 8), 256, 0, stream>>>(
            hb, W2t, MM, CC, FF, tiles, /*mode=*/3,
            b2, x3f, (float*)d_out, nullptr, nullptr, nullptr, nullptr);
    }

    (void)in_sizes; (void)n_in; (void)out_size; (void)ws_size;
}
